// SwinTransformerBlock_83958020702818
// MI455X (gfx1250) — matrix operand order repaired and checked by reading, not compiled, not run on hardware
//
#include <hip/hip_runtime.h>

typedef __attribute__((ext_vector_type(16))) __bf16 v16bf;
typedef __attribute__((ext_vector_type(8)))  __bf16 v8bf;
typedef __attribute__((ext_vector_type(8)))  float  v8f;
typedef unsigned int u32x4 __attribute__((ext_vector_type(4)));
typedef int          i32x4 __attribute__((ext_vector_type(4)));
typedef int          i32x8 __attribute__((ext_vector_type(8)));

#define TOK   100352   // B*H*W = 32*56*56
#define CCH   384
#define NHEAD 12
#define HDIM  32
#define BNT   2048     // B*NW = 32*64
#define LWIN  49

#if defined(__AMDGCN__) && __has_builtin(__builtin_amdgcn_tensor_load_to_lds) && \
    __has_builtin(__builtin_amdgcn_s_wait_tensorcnt)
#define USE_TDM 1
#else
#define USE_TDM 0
#endif

#if __has_include(<hip/amd_detail/amd_gfx1250_TDM.h>)
#define TDM_6ARG 1
#else
#define TDM_6ARG 0
#endif

// ---------- helpers ----------
__device__ __forceinline__ unsigned short f2bf(float x) {
  union { float f; unsigned u; } v; v.f = x;
  unsigned r = v.u + 0x7FFFu + ((v.u >> 16) & 1u);
  return (unsigned short)(r >> 16);
}

// DPP row rotate (16-lane row) -- keeps reductions in the VALU pipe (no ds_bpermute)
template<int CTRL>
__device__ __forceinline__ float dpp_rot(float x) {
  int r = __builtin_amdgcn_update_dpp(0, __builtin_bit_cast(int, x), CTRL, 0xF, 0xF, true);
  return __builtin_bit_cast(float, r);
}
__device__ __forceinline__ float rowmax16(float x) {
  x = fmaxf(x, dpp_rot<0x121>(x));  // row_ror:1
  x = fmaxf(x, dpp_rot<0x122>(x));  // row_ror:2
  x = fmaxf(x, dpp_rot<0x124>(x));  // row_ror:4
  x = fmaxf(x, dpp_rot<0x128>(x));  // row_ror:8
  return x;
}
__device__ __forceinline__ float rowsum16(float x) {
  x += dpp_rot<0x121>(x);
  x += dpp_rot<0x122>(x);
  x += dpp_rot<0x124>(x);
  x += dpp_rot<0x128>(x);
  return x;
}

// A fragment (16x32 bf16): lane holds row m=lane&15; e=0..7 -> K=g*8+e, e=8..15 -> K=16+g*8+(e-8)
__device__ __forceinline__ v16bf frag_a(const unsigned short* tile, int row0, int stride, int k0, int lane) {
  int m = lane & 15, g = (lane >> 4) & 1;
  const unsigned short* p = tile + (size_t)(row0 + m) * stride + k0 + g * 8;
  union { v16bf v; v8bf h[2]; } u;
  u.h[0] = *(const v8bf*)p;
  u.h[1] = *(const v8bf*)(p + 16);
  return u.v;
}

// B fragment (32x16 bf16, stored [N][K]): lane holds col n=lane&15; e=0..7 -> K=g*8+e, e=8..15 -> K=16+g*8+(e-8)
__device__ __forceinline__ v16bf frag_b(const unsigned short* tile, int row0, int stride, int k0, int lane) {
  int n = lane & 15, g = (lane >> 4) & 1;
  const unsigned short* p = tile + (size_t)(row0 + n) * stride + k0 + g * 8;
  union { v16bf v; v8bf h[2]; } u;
  u.h[0] = *(const v8bf*)p;
  u.h[1] = *(const v8bf*)(p + 16);
  return u.v;
}

__device__ __forceinline__ v8f wmma_bf16(v16bf a, v16bf b, v8f c) {
  return __builtin_amdgcn_wmma_f32_16x16x32_bf16(false, a, false, b, (short)0, c, false, false);
}

#if USE_TDM
// Issue a TDM 2D tile load: tile_dim0=32 elems (64B) x tile_dim1=128 rows, bf16,
// row stride = K elems; LDS padding 16B after every 64B -> padded row stride 80B (40 shorts),
// matching the frag_a/frag_b LDS layout.
__device__ __forceinline__ void tdm_load_tile(unsigned lds_off, const unsigned short* gptr,
                                              int K, int td0_rem, int td1_rem) {
  unsigned long long ga = (unsigned long long)gptr;
  u32x4 g0;
  g0[0] = 1u;                                              // count=1 valid descriptor
  g0[1] = lds_off;                                         // LDS byte address
  g0[2] = (unsigned)ga;                                    // global_addr[31:0]
  g0[3] = (unsigned)((ga >> 32) & 0x1FFFFFFu) | (2u << 30); // global_addr[56:32], type=2
  unsigned td0 = (unsigned)td0_rem;                        // remaining elems along dim0
  unsigned td1 = (unsigned)td1_rem;                        // remaining rows
  i32x8 g1;
  g1[0] = (int)((1u << 16) |      // data_size = 2 bytes
                (1u << 20) |      // pad_enable
                (3u << 22) |      // pad_interval: 16 DWORDs (64B)
                (3u << 25));      // pad_amount: 4 DWORDs (16B)
  g1[1] = (int)((td0 & 0xFFFFu) << 16);                    // tensor_dim0[15:0] @ bits 63:48
  g1[2] = (int)(((td0 >> 16) & 0xFFFFu) | ((td1 & 0xFFFFu) << 16)); // td0 hi, td1 lo
  g1[3] = (int)(((td1 >> 16) & 0xFFFFu) | (32u << 16));    // td1 hi, tile_dim0 = 32
  g1[4] = 128;                                             // tile_dim1 = 128, tile_dim2 = 0
  g1[5] = (int)(unsigned)K;                                // tensor_dim0_stride[31:0]
  g1[6] = 0;                                               // stride hi, dim1_stride lo
  g1[7] = 0;
  i32x4 gz = {0, 0, 0, 0};
#if TDM_6ARG
  i32x8 gz8 = {0, 0, 0, 0, 0, 0, 0, 0};
  __builtin_amdgcn_tensor_load_to_lds(g0, g1, gz, gz, gz8, 0);
#else
  __builtin_amdgcn_tensor_load_to_lds(g0, g1, gz, gz, 0);
#endif
}
#endif

// ---------- f32 -> bf16 convert ----------
__global__ __launch_bounds__(256) void cvt_kernel(const float* __restrict__ src,
                                                  unsigned short* __restrict__ dst, int n) {
  int i = blockIdx.x * 256 + threadIdx.x;
  if (i < n) dst[i] = f2bf(src[i]);
}

// ---------- LayerNorm (one wave per token), optionally fused shift + window partition ----------
template<bool SHIFTWIN>
__global__ __launch_bounds__(256) void ln_kernel(const float* __restrict__ xin,
                                                 const float* __restrict__ gam,
                                                 const float* __restrict__ bet,
                                                 unsigned short* __restrict__ out) {
  int token = blockIdx.x * 8 + (threadIdx.x >> 5);
  int lane  = threadIdx.x & 31;
  const float* xr = xin + (size_t)token * CCH;
  float vals[12]; float s = 0.f, s2 = 0.f;
#pragma unroll
  for (int t = 0; t < 12; ++t) { float v = xr[t * 32 + lane]; vals[t] = v; s += v; s2 += v * v; }
#pragma unroll
  for (int off = 1; off < 32; off <<= 1) { s += __shfl_xor(s, off, 32); s2 += __shfl_xor(s2, off, 32); }
  float mean = s * (1.f / 384.f);
  float var  = s2 * (1.f / 384.f) - mean * mean;
  float rstd = rsqrtf(var + 1e-5f);
  size_t orow;
  if (SHIFTWIN) {
    int b = token / 3136, pix = token % 3136;
    int hh = pix / 56, ww = pix % 56;
    int hs = hh >= 3 ? hh - 3 : hh + 53;   // shifted coords: roll(-3)
    int wsx = ww >= 3 ? ww - 3 : ww + 53;
    int row = ((b * 8 + hs / 7) * 8 + wsx / 7) * LWIN + (hs % 7) * 7 + (wsx % 7);
    orow = (size_t)row * CCH;
  } else {
    orow = (size_t)token * CCH;
  }
#pragma unroll
  for (int t = 0; t < 12; ++t) {
    int c = t * 32 + lane;
    out[orow + c] = f2bf((vals[t] - mean) * rstd * gam[c] + bet[c]);
  }
}

// ---------- precompute (relpos bias + shift mask) per (win, head), padded 64x64 ----------
__global__ __launch_bounds__(256) void bias_kernel(const float* __restrict__ amask,
                                                   const int* __restrict__ rpi,
                                                   const float* __restrict__ rpb,
                                                   float* __restrict__ biasbuf) {
  int wh = blockIdx.x;            // win*12 + head, 0..767
  int win = wh / 12, head = wh % 12;
  float* o = biasbuf + (size_t)wh * 4096;
  for (int e = threadIdx.x; e < 4096; e += 256) {
    int m = e >> 6, n = e & 63;
    float v = 0.f;
    if (m < LWIN && n < LWIN) {
      int idx = rpi[m * LWIN + n];
      v = rpb[idx * NHEAD + head] + amask[(size_t)win * 2401 + m * LWIN + n];
    }
    o[e] = v;
  }
}

// ---------- generic WMMA GEMM: D = A[R][K] * Bw[N][K]^T, 128x128 block tile ----------
// TDM double-buffered tile streaming when available; cooperative loads otherwise.
// EPI 0: QKV scatter (+bias, q*scale) -> bf16   1: proj (+bias, window reverse, +resid) -> f32
//     2: fc1 (+bias, GELU) -> bf16              3: fc2 (+bias, += outf in place) -> f32
template<int EPI>
__global__ __launch_bounds__(256) void gemm_wmma_k(const unsigned short* __restrict__ A,
                                                   const unsigned short* __restrict__ Bw,
                                                   const float* __restrict__ bias,
                                                   unsigned short* __restrict__ outb,
                                                   float* __restrict__ outf,
                                                   const float* __restrict__ resid,
                                                   int K, int N, int R) {
  __shared__ alignas(16) unsigned short As[2][128 * 40];
  __shared__ alignas(16) unsigned short Bs[2][128 * 40];
  const int tid  = threadIdx.x;
  const int lane = tid & 31;
  const int wave = tid >> 5;
  const int wm = (wave >> 2) * 64;
  const int wn = (wave & 3) * 32;
  const int r0 = blockIdx.y * 128;
  const int n0 = blockIdx.x * 128;
  const int nsteps = K >> 5;

  v8f acc[4][2] = {};

#if USE_TDM
  if (wave == 0) {
    tdm_load_tile((unsigned)(size_t)&As[0][0], &A[(size_t)r0 * K], K, K, R - r0);
    tdm_load_tile((unsigned)(size_t)&Bs[0][0], &Bw[(size_t)n0 * K], K, K, N - n0);
    if (nsteps > 1) {
      tdm_load_tile((unsigned)(size_t)&As[1][0], &A[(size_t)r0 * K + 32], K, K - 32, R - r0);
      tdm_load_tile((unsigned)(size_t)&Bs[1][0], &Bw[(size_t)n0 * K + 32], K, K - 32, N - n0);
    }
  }
  for (int i = 0; i < nsteps; ++i) {
    const int buf = i & 1;
    if (wave == 0) {
      if (i + 1 < nsteps) __builtin_amdgcn_s_wait_tensorcnt(2);   // pair i complete
      else               __builtin_amdgcn_s_wait_tensorcnt(0);
    }
    __syncthreads();
    v16bf a[4];
#pragma unroll
    for (int mt = 0; mt < 4; ++mt) a[mt] = frag_a(As[buf], wm + mt * 16, 40, 0, lane);
#pragma unroll
    for (int nt = 0; nt < 2; ++nt) {
      v16bf b = frag_b(Bs[buf], wn + nt * 16, 40, 0, lane);
#pragma unroll
      for (int mt = 0; mt < 4; ++mt) acc[mt][nt] = wmma_bf16(a[mt], b, acc[mt][nt]);
    }
    __syncthreads();
    if (wave == 0 && i + 2 < nsteps) {
      int k0 = (i + 2) * 32;
      tdm_load_tile((unsigned)(size_t)&As[buf][0], &A[(size_t)r0 * K + k0], K, K - k0, R - r0);
      tdm_load_tile((unsigned)(size_t)&Bs[buf][0], &Bw[(size_t)n0 * K + k0], K, K - k0, N - n0);
    }
  }
#else
  for (int k0 = 0; k0 < K; k0 += 32) {
#pragma unroll
    for (int i = 0; i < 2; ++i) {
      int idx = tid + i * 256;
      int row = idx >> 2, cg = (idx & 3) * 8;
      *(v8bf*)&As[0][row * 40 + cg] = *(const v8bf*)&A[(size_t)(r0 + row) * K + k0 + cg];
      *(v8bf*)&Bs[0][row * 40 + cg] = *(const v8bf*)&Bw[(size_t)(n0 + row) * K + k0 + cg];
      if (k0 + 32 < K) __builtin_prefetch(&A[(size_t)(r0 + row) * K + k0 + 32 + cg], 0, 0);
    }
    __syncthreads();
    v16bf a[4];
#pragma unroll
    for (int mt = 0; mt < 4; ++mt) a[mt] = frag_a(As[0], wm + mt * 16, 40, 0, lane);
#pragma unroll
    for (int nt = 0; nt < 2; ++nt) {
      v16bf b = frag_b(Bs[0], wn + nt * 16, 40, 0, lane);
#pragma unroll
      for (int mt = 0; mt < 4; ++mt) acc[mt][nt] = wmma_bf16(a[mt], b, acc[mt][nt]);
    }
    __syncthreads();
  }
#endif

  const int lg = lane >> 4, nl = lane & 15;
#pragma unroll
  for (int mt = 0; mt < 4; ++mt) {
#pragma unroll
    for (int nt = 0; nt < 2; ++nt) {
      int n = n0 + wn + nt * 16 + nl;
      float bv = bias[n];
#pragma unroll
      for (int v = 0; v < 8; ++v) {
        int r = r0 + wm + mt * 16 + lg * 8 + v;
        float val = acc[mt][nt][v] + bv;
        if (EPI == 0) {
          int three = n / 384, rem = n % 384;
          int head = rem >> 5, hd = rem & 31;
          if (three == 0) val *= 0.1767766953f;  // HD^-0.5 folded into q
          int Bn = r / LWIN, l = r % LWIN;
          outb[(size_t)(((three * BNT + Bn) * NHEAD + head) * LWIN + l) * HDIM + hd] = f2bf(val);
        } else if (EPI == 1) {
          int Bn = r / LWIN, l = r % LWIN;
          int b = Bn >> 6, win = Bn & 63;
          int hh_s = (win >> 3) * 7 + l / 7;
          int ww_s = (win & 7) * 7 + l % 7;
          int hh = hh_s < 53 ? hh_s + 3 : hh_s - 53;  // unshift: roll(+3)
          int ww = ww_s < 53 ? ww_s + 3 : ww_s - 53;
          size_t tok = (size_t)b * 3136 + hh * 56 + ww;
          outf[tok * CCH + n] = resid[tok * CCH + n] + val;
        } else if (EPI == 2) {
          val = 0.5f * val * (1.0f + erff(val * 0.7071067812f));  // exact GELU
          outb[(size_t)r * N + n] = f2bf(val);
        } else {  // EPI == 3
          size_t o = (size_t)r * CCH + n;
          outf[o] = outf[o] + val;
        }
      }
    }
  }
}

// ---------- attention: one block (4 waves) per (window, head) ----------
__global__ __launch_bounds__(128) void attn_kernel(const unsigned short* __restrict__ qkv,
                                                   const float* __restrict__ biasbuf,
                                                   unsigned short* __restrict__ ao) {
  int bh = blockIdx.x;             // 0 .. 24575
  int Bn = bh / NHEAD, head = bh % NHEAD;
  int win = Bn & 63;

  __shared__ alignas(16) unsigned short qs[64 * 40];
  __shared__ alignas(16) unsigned short ks[64 * 40];
  __shared__ alignas(16) unsigned short vt[32 * 72];
  __shared__ alignas(16) unsigned short ps[64 * 72];

  const size_t hslab = (size_t)BNT * NHEAD * (LWIN * HDIM);
  const unsigned short* Qg = qkv + (size_t)(Bn * NHEAD + head) * (LWIN * HDIM);
  const unsigned short* Kg = Qg + hslab;
  const unsigned short* Vg = Kg + hslab;

  int tid = threadIdx.x;
  for (int e = tid; e < 64 * 32; e += 128) {
    int l = e >> 5, hd = e & 31;
    unsigned short qv = 0, kv = 0;
    if (l < LWIN) { qv = Qg[l * 32 + hd]; kv = Kg[l * 32 + hd]; }
    qs[l * 40 + hd] = qv;
    ks[l * 40 + hd] = kv;
  }
  for (int e = tid; e < 32 * 64; e += 128) {
    int hd = e >> 6, l = e & 63;
    vt[hd * 72 + l] = (l < LWIN) ? Vg[l * 32 + hd] : (unsigned short)0;
  }
  __syncthreads();

  int wave = tid >> 5, lane = tid & 31;
  int lg = lane >> 4, nl = lane & 15;
  int mbase = wave * 16;

  // S = q k^T  (16x64 per wave; K = HD = 32 -> one WMMA per column tile)
  v16bf aq = frag_a(qs, mbase, 40, 0, lane);
  v8f s[4] = {};
#pragma unroll
  for (int nt = 0; nt < 4; ++nt) {
    v16bf bk = frag_b(ks, nt * 16, 40, 0, lane);
    s[nt] = wmma_bf16(aq, bk, s[nt]);
  }

  // + relpos bias + shift mask
  const float* bb = biasbuf + ((size_t)win * NHEAD + head) * 4096;
#pragma unroll
  for (int nt = 0; nt < 4; ++nt) {
    int n = nt * 16 + nl;
#pragma unroll
    for (int v = 0; v < 8; ++v) {
      int m = mbase + lg * 8 + v;
      s[nt][v] += bb[m * 64 + n];
    }
  }

  // softmax over n=0..48 (pad cols live in tile 3, nl>=1); row = fixed (v, half-wave)
  // reductions use DPP row_ror rotations: pure VALU, no LDS round trips
#pragma unroll
  for (int v = 0; v < 8; ++v) {
    float mx = -3.0e38f;
#pragma unroll
    for (int nt = 0; nt < 4; ++nt) {
      float xv = s[nt][v];
      if (nt == 3 && nl >= 1) xv = -3.0e38f;
      mx = fmaxf(mx, xv);
    }
    mx = rowmax16(mx);
    float sum = 0.f;
#pragma unroll
    for (int nt = 0; nt < 4; ++nt) {
      float ev = __expf(s[nt][v] - mx);
      if (nt == 3 && nl >= 1) ev = 0.f;
      s[nt][v] = ev;
      sum += ev;
    }
    sum = rowsum16(sum);
    float rinv = 1.0f / sum;
#pragma unroll
    for (int nt = 0; nt < 4; ++nt) s[nt][v] *= rinv;
  }

  // stage P (bf16) to LDS in A-matrix order
#pragma unroll
  for (int nt = 0; nt < 4; ++nt) {
    int n = nt * 16 + nl;
#pragma unroll
    for (int v = 0; v < 8; ++v) {
      int m = mbase + lg * 8 + v;
      ps[m * 72 + n] = f2bf(s[nt][v]);
    }
  }
  __syncthreads();

  // O = P @ V  (K = 64 -> 2 WMMAs per 16-wide hd tile)
  v16bf a0 = frag_a(ps, mbase, 72, 0, lane);
  v16bf a1 = frag_a(ps, mbase, 72, 32, lane);
#pragma unroll
  for (int nt2 = 0; nt2 < 2; ++nt2) {
    v16bf b0 = frag_b(vt, nt2 * 16, 72, 0, lane);
    v16bf b1 = frag_b(vt, nt2 * 16, 72, 32, lane);
    v8f o = {};
    o = wmma_bf16(a0, b0, o);
    o = wmma_bf16(a1, b1, o);
#pragma unroll
    for (int v = 0; v < 8; ++v) {
      int m = mbase + lg * 8 + v;
      if (m < LWIN) {
        size_t r = (size_t)Bn * LWIN + m;
        ao[r * CCH + head * HDIM + nt2 * 16 + nl] = f2bf(o[v]);
      }
    }
  }
}

// ---------- host-side orchestration ----------
extern "C" void kernel_launch(void* const* d_in, const int* in_sizes, int n_in,
                              void* d_out, int out_size, void* d_ws, size_t ws_size,
                              hipStream_t stream) {
  const float* x      = (const float*)d_in[0];
  const float* amask  = (const float*)d_in[1];
  const int*   rpi    = (const int*)d_in[2];
  const float* n1g    = (const float*)d_in[3];
  const float* n1b    = (const float*)d_in[4];
  const float* qkv_w  = (const float*)d_in[5];
  const float* qkv_b  = (const float*)d_in[6];
  const float* rpb    = (const float*)d_in[7];
  const float* proj_w = (const float*)d_in[8];
  const float* proj_b = (const float*)d_in[9];
  const float* n2g    = (const float*)d_in[10];
  const float* n2b    = (const float*)d_in[11];
  const float* fc1_w  = (const float*)d_in[12];
  const float* fc1_b  = (const float*)d_in[13];
  const float* fc2_w  = (const float*)d_in[14];
  const float* fc2_b  = (const float*)d_in[15];
  float* out = (float*)d_out;

  char* ws = (char*)d_ws;
  size_t off = 0;
  auto take = [&](size_t bytes) -> char* {
    char* p = ws + off;
    off = (off + bytes + 255) & ~(size_t)255;
    return p;
  };
  unsigned short* wqkv = (unsigned short*)take((size_t)1152 * 384 * 2);
  unsigned short* wprj = (unsigned short*)take((size_t)384 * 384 * 2);
  unsigned short* wfc1 = (unsigned short*)take((size_t)1536 * 384 * 2);
  unsigned short* wfc2 = (unsigned short*)take((size_t)384 * 1536 * 2);
  unsigned short* hb   = (unsigned short*)take((size_t)TOK * CCH * 2);   // LN1 out; reused for LN2 out
  unsigned short* big  = (unsigned short*)take((size_t)TOK * 1536 * 2);  // qkv, then fc1 out
  float*          bbuf = (float*)take((size_t)64 * NHEAD * 4096 * 4);
  unsigned short* ao   = (unsigned short*)take((size_t)TOK * CCH * 2);

  // 1) weights -> bf16
  cvt_kernel<<<(1152 * 384 + 255) / 256, 256, 0, stream>>>(qkv_w, wqkv, 1152 * 384);
  cvt_kernel<<<(384 * 384 + 255) / 256, 256, 0, stream>>>(proj_w, wprj, 384 * 384);
  cvt_kernel<<<(1536 * 384 + 255) / 256, 256, 0, stream>>>(fc1_w, wfc1, 1536 * 384);
  cvt_kernel<<<(384 * 1536 + 255) / 256, 256, 0, stream>>>(fc2_w, wfc2, 384 * 1536);

  // 2) LN1 + shift + window partition -> hb (bf16, windowed row order)
  ln_kernel<true><<<TOK / 8, 256, 0, stream>>>(x, n1g, n1b, hb);

  // 3) bias table per (window, head)
  bias_kernel<<<64 * NHEAD, 256, 0, stream>>>(amask, rpi, rpb, bbuf);

  // 4) QKV GEMM -> scattered q/k/v (q pre-scaled)
  gemm_wmma_k<0><<<dim3(1152 / 128, TOK / 128), 256, 0, stream>>>(
      hb, wqkv, qkv_b, big, nullptr, nullptr, 384, 1152, TOK);

  // 5) attention per (window, head)
  attn_kernel<<<BNT * NHEAD, 128, 0, stream>>>(big, bbuf, ao);

  // 6) proj GEMM + window reverse + unshift + residual -> d_out (f32)
  gemm_wmma_k<1><<<dim3(384 / 128, TOK / 128), 256, 0, stream>>>(
      ao, wprj, proj_b, nullptr, out, x, 384, 384, TOK);

  // 7) LN2 -> bf16 (reuse hb region)
  ln_kernel<false><<<TOK / 8, 256, 0, stream>>>(out, n2g, n2b, hb);

  // 8) fc1 GEMM + GELU -> big (bf16)
  gemm_wmma_k<2><<<dim3(1536 / 128, TOK / 128), 256, 0, stream>>>(
      hb, wfc1, fc1_b, big, nullptr, nullptr, 384, 1536, TOK);

  // 9) fc2 GEMM + residual (in-place on d_out)
  gemm_wmma_k<3><<<dim3(384 / 128, TOK / 128), 256, 0, stream>>>(
      big, wfc2, fc2_b, nullptr, out, nullptr, 1536, 384, TOK);
}
